// AttentiveGraphConvolution_9972914061610
// MI455X (gfx1250) — matrix operand order repaired and checked by reading, not compiled, not run on hardware
//
#include <hip/hip_runtime.h>
#include <hip/hip_bf16.h>

typedef __attribute__((ext_vector_type(16))) __bf16 v16bf;
typedef __attribute__((ext_vector_type(8)))  __bf16 v8bf;
typedef __attribute__((ext_vector_type(8)))  float  v8f;

#define N_NODES 8192
#define D_IN    512
#define D_OUT   128
#define NEGBIG  (-1.0e30f)

// ---------------------------------------------------------------------------
// Kernel 0: pre-pack W (f32, 512x128) into WMMA-B-operand-ready bf16 tiles.
// WB layout: [kt(16)][nt(8)][lane(32)][16 bf16]  -> kernel 1 reads one
// contiguous 32B v16bf per lane per tile (global_load_b128 x2, coalesced).
// ---------------------------------------------------------------------------
__global__ void gat_pack_w(const float* __restrict__ W, __bf16* __restrict__ WB) {
    const int tid  = blockIdx.x * blockDim.x + threadIdx.x;   // 4096 threads
    const int tile = tid >> 5;                                // 0..127 = kt*8+nt
    const int lane = tid & 31;
    const int kt = tile >> 3, nt = tile & 7;
    const int hi = lane >> 4, lr = lane & 15;
    __bf16* dst = WB + (((size_t)tile * 32 + lane) << 4);
#pragma unroll
    for (int t = 0; t < 16; ++t)
        dst[t] = (__bf16)W[(size_t)(kt * 32 + (t & 7) + 8 * hi + 16 * (t >> 3)) * D_OUT + nt * 16 + lr];
}

// ---------------------------------------------------------------------------
// Kernel 1: h = x @ W (bf16 WMMA, f32 acc). One wave = 16-row x 128-col strip.
// Also computes a_s = h.attn_self, a_n = h.attn_neigh in-wave (shfl_xor
// reduction), and stores H^T bf16 column-major with packed 16B stores.
// ---------------------------------------------------------------------------
__global__ void gat_gemm_h(const float* __restrict__ x, const __bf16* __restrict__ WB,
                           const float* __restrict__ ws_, const float* __restrict__ wn_,
                           __bf16* __restrict__ HT,
                           float* __restrict__ a_s, float* __restrict__ a_n) {
    const int wave = (blockIdx.x * blockDim.x + threadIdx.x) >> 5;  // 0..511 exact
    const int lane = threadIdx.x & 31;
    const int hi   = lane >> 4;
    const int lr   = lane & 15;
    const int rb   = wave;
    const int arow = rb * 16 + lr;

    v8f c[8];
#pragma unroll
    for (int nt = 0; nt < 8; ++nt) c[nt] = (v8f){0.f,0.f,0.f,0.f,0.f,0.f,0.f,0.f};

    const float4* xr = (const float4*)(x + (size_t)arow * D_IN);

    for (int kt = 0; kt < D_IN / 32; ++kt) {
        const int k0 = kt * 32;
        // A lane data: two contiguous 8-float runs -> 4 float4 loads
        const int b = (k0 + 8 * hi) >> 2;
        const float4 X0 = xr[b];
        const float4 X1 = xr[b + 1];
        const float4 X2 = xr[b + 4];
        const float4 X3 = xr[b + 5];
        const float xa[16] = {X0.x,X0.y,X0.z,X0.w, X1.x,X1.y,X1.z,X1.w,
                              X2.x,X2.y,X2.z,X2.w, X3.x,X3.y,X3.z,X3.w};
        v16bf Aa;
#pragma unroll
        for (int i = 0; i < 16; ++i) Aa[i] = (__bf16)xa[i];

#pragma unroll
        for (int nt = 0; nt < 8; ++nt) {
            const v16bf Bb = *(const v16bf*)(WB + (((size_t)(kt * 8 + nt) * 32 + lane) << 4));
            c[nt] = __builtin_amdgcn_wmma_f32_16x16x32_bf16(
                false, Aa, false, Bb, (short)0, c[nt], false, false);
        }
    }

    // --- store H^T (bf16, column-major): 8 consecutive rows -> one 16B store
#pragma unroll
    for (int nt = 0; nt < 8; ++nt) {
        v8bf hv;
#pragma unroll
        for (int r = 0; r < 8; ++r) hv[r] = (__bf16)c[nt][r];
        *(v8bf*)(HT + (size_t)(nt * 16 + lr) * N_NODES + rb * 16 + 8 * hi) = hv;
    }

    // --- a_s / a_n: per-lane partial over this lane's 8 columns, then
    // shfl_xor reduce across the 16-lane half (rows constant across lr).
    float wsc[8], wnc[8];
#pragma unroll
    for (int nt = 0; nt < 8; ++nt) {
        wsc[nt] = ws_[nt * 16 + lr];
        wnc[nt] = wn_[nt * 16 + lr];
    }
#pragma unroll
    for (int r = 0; r < 8; ++r) {
        float ps = 0.f, pn = 0.f;
#pragma unroll
        for (int nt = 0; nt < 8; ++nt) {
            ps += c[nt][r] * wsc[nt];
            pn += c[nt][r] * wnc[nt];
        }
#pragma unroll
        for (int off = 1; off < 16; off <<= 1) {
            ps += __shfl_xor(ps, off);
            pn += __shfl_xor(pn, off);
        }
        if (lr == 0) {
            const int row = rb * 16 + r + 8 * hi;
            a_s[row] = ps;
            a_n[row] = pn;
        }
    }
}

// ---------------------------------------------------------------------------
// Kernel 3: fused masked-softmax attention + P@H, flash style, 2-way
// column split per 16-row block with LDS merge of (m, l, O) partials.
// Block = 128 threads = 2 row-blocks x 2 column-halves.
// LDS staging is per-lane contiguous: writer lane (nt,r) == reader lane
// (nt,r), so partials move as 32B v8f ds ops instead of scalar b32.
// ---------------------------------------------------------------------------
__global__ void gat_attn_out(const float* __restrict__ adj, const __bf16* __restrict__ HT,
                             const float* __restrict__ a_s, const float* __restrict__ a_n,
                             float* __restrict__ out) {
    __shared__ float ldsP[2][32][64];   // [row-block][lane][nt*8+r] partial O
    __shared__ float ldsM[2][16];
    __shared__ float ldsL[2][16];

    const int wv   = threadIdx.x >> 5;     // 0..3
    const int rbl  = wv >> 1;              // row-block within block
    const int half = wv & 1;               // column half
    const int lane = threadIdx.x & 31;
    const int hi   = lane >> 4;
    const int lr   = lane & 15;
    const int rb   = blockIdx.x * 2 + rbl; // 0..511 exact

    const int   erow = rb * 16 + lr;
    const float asv  = a_s[erow];

    float m = NEGBIG, lsum = 0.f;
    v8f c[8];
#pragma unroll
    for (int nt = 0; nt < 8; ++nt) c[nt] = (v8f){0.f,0.f,0.f,0.f,0.f,0.f,0.f,0.f};

    const float4* adjr = (const float4*)(adj + (size_t)erow * N_NODES);

    // interleaved 32-col tiles: ct = 2*t + half
    for (int t = 0; t < N_NODES / 64; ++t) {
        const int col0 = (2 * t + half) * 32;
        const int f4b  = col0 >> 2;

        // prefetch the adjacency stream 4 iterations (256 cols) ahead
        const int colpf = col0 + 256;
        if (colpf < N_NODES)
            __builtin_prefetch((const void*)(adj + (size_t)erow * N_NODES + colpf + hi * 16), 0, 0);

        const float4 A0 = adjr[f4b + 4 * hi];
        const float4 A1 = adjr[f4b + 4 * hi + 1];
        const float4 A2 = adjr[f4b + 4 * hi + 2];
        const float4 A3 = adjr[f4b + 4 * hi + 3];
        const float4* an4 = (const float4*)(a_n + col0);
        const float4 N0 = an4[4 * hi];
        const float4 N1 = an4[4 * hi + 1];
        const float4 N2 = an4[4 * hi + 2];
        const float4 N3 = an4[4 * hi + 3];

        const float av[16] = {A0.x,A0.y,A0.z,A0.w, A1.x,A1.y,A1.z,A1.w,
                              A2.x,A2.y,A2.z,A2.w, A3.x,A3.y,A3.z,A3.w};
        const float nv[16] = {N0.x,N0.y,N0.z,N0.w, N1.x,N1.y,N1.z,N1.w,
                              N2.x,N2.y,N2.z,N2.w, N3.x,N3.y,N3.z,N3.w};

        float ev[16];
        float lmax = NEGBIG;
#pragma unroll
        for (int i = 0; i < 16; ++i) {
            float e = asv + nv[i];
            e = (e >= 0.f) ? e : 0.2f * e;          // leaky_relu(0.2)
            ev[i]  = (av[i] != 0.f) ? e : NEGBIG;   // mask non-edges
            lmax   = fmaxf(lmax, ev[i]);
        }
        const float tmax = fmaxf(lmax, __shfl_xor(lmax, 16));

        // sparse skip: whole 16x32 tile has no edges (wave-uniform branch)
        if (__ballot(tmax > -1.0e29f) == 0ull) continue;

        const float newm  = fmaxf(m, tmax);
        const float scale = __expf(m - newm);
        m = newm;

        v16bf Pa;
        float ps = 0.f;
#pragma unroll
        for (int i = 0; i < 16; ++i) {
            const float p = (ev[i] > -1.0e29f) ? __expf(ev[i] - m) : 0.f;
            ps += p;
            Pa[i] = (__bf16)p;
        }
        ps += __shfl_xor(ps, 16);
        lsum = lsum * scale + ps;

        // rescale accumulators: VGPR r belongs to row (r + 8*hi) whose scale
        // lives in lane (r + 8*hi) -> per-lane-src shfl (ds_bpermute)
        v8f scv;
#pragma unroll
        for (int r = 0; r < 8; ++r) scv[r] = __shfl(scale, r + hi * 8);
#pragma unroll
        for (int nt = 0; nt < 8; ++nt) c[nt] *= scv;

        // P(16x32) @ H(32x128): 8 bf16 WMMAs, B contiguous from H^T
#pragma unroll
        for (int nt = 0; nt < 8; ++nt) {
            const v16bf Bb = *(const v16bf*)(HT + (size_t)(nt * 16 + lr) * N_NODES + col0 + hi * 16);
            c[nt] = __builtin_amdgcn_wmma_f32_16x16x32_bf16(
                false, Pa, false, Bb, (short)0, c[nt], false, false);
        }
    }

    // -------- merge the two column halves through LDS (vector ds ops) ------
    if (half == 1) {
        float* dst = &ldsP[rbl][lane][0];
#pragma unroll
        for (int nt = 0; nt < 8; ++nt)
            *(v8f*)(dst + nt * 8) = c[nt];
        if (hi == 0) {
            ldsM[rbl][lr] = m;
            ldsL[rbl][lr] = lsum;
        }
    }
    __syncthreads();
    if (half != 0) return;

    const float m2 = ldsM[rbl][lr];
    const float l2 = ldsL[rbl][lr];
    const float M  = fmaxf(m, m2);
    const float f1 = __expf(m - M);       // 0 if this half saw no edges
    const float f2 = __expf(m2 - M);
    const float L  = lsum * f1 + l2 * f2; // > 0: every row has a self-loop

    v8f f1v, f2v, liv;
#pragma unroll
    for (int r = 0; r < 8; ++r) {
        f1v[r] = __shfl(f1, r + hi * 8);
        f2v[r] = __shfl(f2, r + hi * 8);
        liv[r] = 1.0f / __shfl(L, r + hi * 8);
    }

    const float* src = &ldsP[rbl][lane][0];
#pragma unroll
    for (int nt = 0; nt < 8; ++nt) {
        const v8f o2  = *(const v8f*)(src + nt * 8);
        const int col = nt * 16 + lr;
#pragma unroll
        for (int r = 0; r < 8; ++r) {
            const float v = (c[nt][r] * f1v[r] + o2[r] * f2v[r]) * liv[r];
            out[(size_t)(rb * 16 + r + 8 * hi) * D_OUT + col] = (v > 0.f) ? v : 0.f;
        }
    }
}

// ---------------------------------------------------------------------------
extern "C" void kernel_launch(void* const* d_in, const int* in_sizes, int n_in,
                              void* d_out, int out_size, void* d_ws, size_t ws_size,
                              hipStream_t stream) {
    const float* x      = (const float*)d_in[0];
    const float* adj    = (const float*)d_in[1];
    const float* W      = (const float*)d_in[2];
    const float* attn_s = (const float*)d_in[3];
    const float* attn_n = (const float*)d_in[4];
    float*       out    = (float*)d_out;

    char* ws = (char*)d_ws;
    __bf16* HT = (__bf16*)ws;                                   // 2 MB
    size_t off = (size_t)D_OUT * N_NODES * sizeof(__bf16);
    __bf16* WB = (__bf16*)(ws + off);                           // 128 KB
    off += (size_t)D_IN * D_OUT * sizeof(__bf16);
    float* a_s = (float*)(ws + off);                            // 32 KB
    off += (size_t)N_NODES * sizeof(float);
    float* a_n = (float*)(ws + off);                            // 32 KB

    gat_pack_w<<<16, 256, 0, stream>>>(W, WB);
    gat_gemm_h<<<128, 128, 0, stream>>>(x, WB, attn_s, attn_n, HT, a_s, a_n);
    // 512 row-blocks x 2 column-halves = 1024 waves
    gat_attn_out<<<256, 128, 0, stream>>>(adj, HT, a_s, a_n, out);
}
